// FiniteStateGraphAutomaton_36687610642510
// MI455X (gfx1250) — matrix operand order repaired and checked by reading, not compiled, not run on hardware
//
#include <hip/hip_runtime.h>
#include <hip/hip_bf16.h>
#include <math.h>

// FSM graph automaton: per step, the heavy op is a 2048x768x256 GEMM
//   moved[(e,i,u),k] = sum_{t,j} tmp[e,i,t,u,j] * a_norm[t,j,k]
// done with v_wmma_f32_16x16x32_bf16 (bf16 operands, f32 accumulate).
// Everything (TMP 3MB, a_norm 0.4MB bf16, p_move 12.6MB, dist 2MB) is
// L2-resident (192MB) on MI455X; the 48-step chain is latency-bound, so we
// fuse each step into 2 kernels: expand (VALU, per-j s-contraction + accept +
// back reduction) and gemm (WMMA + restart epilogue).

#define NN 256          // nodes
#define SS 4            // FSM states
#define TT 3            // edge types
#define EE 2            // out edges
#define VV 4            // variants
#define NACT 15         // T*S+3
#define STEPS 48
#define MROWS 2048      // E*N*S
#define KDIM 768        // T*N

typedef __attribute__((ext_vector_type(16))) __bf16 v16bf;
typedef __attribute__((ext_vector_type(8)))  float  v8f;

__device__ __forceinline__ unsigned short f2bf(float f) {
  union { float f; unsigned u; } x; x.f = f;
  unsigned r = x.u + 0x7FFFu + ((x.u >> 16) & 1u);   // round-to-nearest-even
  return (unsigned short)(r >> 16);
}

// ---- A1: a_norm, stored transposed bf16: AnT[n][t*256+j] = adj[t][j][n]/max(deg,1)
__global__ __launch_bounds__(256) void fsm_anorm(const float* __restrict__ adj,
                                                 unsigned short* __restrict__ AnT) {
  int tj = blockIdx.x;      // 0..767 : (t*256 + j)
  int n  = threadIdx.x;     // 0..255
  float v = adj[(size_t)tj * NN + n];
  __shared__ float red[256];
  red[n] = v; __syncthreads();
  for (int st = 128; st > 0; st >>= 1) { if (n < st) red[n] += red[n + st]; __syncthreads(); }
  float an = v / fmaxf(red[0], 1.0f);
  AnT[(size_t)n * KDIM + tj] = f2bf(an);
}

// ---- A2: softmax over last axis of log_routing_params (4,8,4,15) -> policy f32
__global__ __launch_bounds__(128) void fsm_policy(const float* __restrict__ lrp,
                                                  float* __restrict__ policy) {
  int r = threadIdx.x;                      // 128 rows of 15
  const float* in = lrp + r * NACT;
  float* out = policy + r * NACT;
  float mx = in[0];
  for (int a = 1; a < NACT; ++a) mx = fmaxf(mx, in[a]);
  float ex[NACT], sum = 0.f;
  for (int a = 0; a < NACT; ++a) { ex[a] = __expf(in[a] - mx); sum += ex[a]; }
  float inv = 1.0f / sum;
  for (int a = 0; a < NACT; ++a) out[a] = ex[a] * inv;
}

// ---- A3: mixed[i,j,s,a] = sum_v vw[i,j,v]*policy[v,nt[j],s,a] -> PM/PA/PB
//      layouts coalesced over j:  PM[(s*12+a)*65536 + i*256 + j], PA/PB[s*65536 + ij]
__global__ __launch_bounds__(256) void fsm_mixed(const float* __restrict__ vw,
                                                 const float* __restrict__ policy,
                                                 const int* __restrict__ ntypes,
                                                 float* __restrict__ PM,
                                                 float* __restrict__ PA,
                                                 float* __restrict__ PB) {
  __shared__ float pol[VV * 8 * SS * NACT];              // 1920 floats
  for (int x = threadIdx.x; x < VV * 8 * SS * NACT; x += 256) pol[x] = policy[x];
  __syncthreads();
  int i = blockIdx.x, j = threadIdx.x;
  int nt = ntypes[j];
  float w[VV];
  #pragma unroll
  for (int v = 0; v < VV; ++v) w[v] = vw[((size_t)i * NN + j) * VV + v];
  int ij = i * NN + j;
  for (int s = 0; s < SS; ++s) {
    for (int a = 0; a < NACT; ++a) {
      float m = 0.f;
      #pragma unroll
      for (int v = 0; v < VV; ++v) m += w[v] * pol[((v * 8 + nt) * SS + s) * NACT + a];
      if (a < TT * SS)           PM[(size_t)(s * 12 + a) * (NN * NN) + ij] = m;
      else if (a == TT * SS)     PA[(size_t)s * (NN * NN) + ij] = m;
      else if (a == TT * SS + 1) PB[(size_t)s * (NN * NN) + ij] = m;
      // a == 14: unused action
    }
  }
}

// ---- A4: dist0[e,i,s,j] = (s==e)*(i==j); accept = 0
__global__ __launch_bounds__(256) void fsm_init(float* __restrict__ dist,
                                                float* __restrict__ accept) {
  int idx = blockIdx.x * 256 + threadIdx.x;   // up to 524288
  if (idx < EE * NN * NN) accept[idx] = 0.f;
  int j = idx & 255, s = (idx >> 8) & 3, i = (idx >> 10) & 255, e = idx >> 18;
  dist[idx] = (s == e && i == j) ? 1.0f : 0.0f;
}

// ---- B1 expand: one block per (e,i) chain, one thread per j.
//  accept[e,i,j] += sum_s dist*PA ; back[e,i] = reduce_j sum_s dist*PB ;
//  TMP[((e*256+i)*4+u)][t*256+j] = bf16( sum_s dist*PM )
__global__ __launch_bounds__(256) void fsm_expand(const float* __restrict__ dist,
                                                  const float* __restrict__ PM,
                                                  const float* __restrict__ PA,
                                                  const float* __restrict__ PB,
                                                  float* __restrict__ accept,
                                                  float* __restrict__ back,
                                                  unsigned short* __restrict__ TMP) {
  int b = blockIdx.x;                 // b = e*256 + i
  int i = b & 255;
  int j = threadIdx.x;
  int ij = i * NN + j;
  float d[SS];
  #pragma unroll
  for (int s = 0; s < SS; ++s) d[s] = dist[((size_t)b * SS + s) * NN + j];

  float acc = 0.f, bk = 0.f;
  #pragma unroll
  for (int s = 0; s < SS; ++s) {
    acc += d[s] * PA[(size_t)s * (NN * NN) + ij];
    bk  += d[s] * PB[(size_t)s * (NN * NN) + ij];
  }
  accept[(size_t)b * NN + j] += acc;

  #pragma unroll
  for (int a = 0; a < TT * SS; ++a) {
    float v = 0.f;
    #pragma unroll
    for (int s = 0; s < SS; ++s) v += d[s] * PM[(size_t)(s * 12 + a) * (NN * NN) + ij];
    int t = a >> 2, u = a & 3;
    TMP[(size_t)(b * SS + u) * KDIM + t * NN + j] = f2bf(v);
  }

  __shared__ float red[256];
  red[j] = bk; __syncthreads();
  for (int st = 128; st > 0; st >>= 1) { if (j < st) red[j] += red[j + st]; __syncthreads(); }
  if (j == 0) back[b] = red[0];
}

// ---- B2 GEMM: dist_new = TMP(2048x768 bf16) @ AnT^T(768x256 bf16) + restart
// One wave (32 threads) per 16x16 tile; K swept in 24 chunks of 32 via
// v_wmma_f32_16x16x32_bf16. VGPR layouts per CDNA5 ISA 7.12.2.
__global__ __launch_bounds__(32) void fsm_gemm(const unsigned short* __restrict__ TMP,
                                               const unsigned short* __restrict__ AnT,
                                               const float* __restrict__ back,
                                               float* __restrict__ dist) {
  int tile = blockIdx.x;        // 0..2047
  int tm = tile >> 4;           // 0..127 (M tiles)
  int tn = tile & 15;           // 0..15  (N tiles)
  int lane = threadIdx.x;
  int h = lane >> 4;            // half-wave select
  int r16 = lane & 15;

  const unsigned* aRow = (const unsigned*)(TMP + (size_t)(tm * 16 + r16) * KDIM);
  const unsigned* bRow = (const unsigned*)(AnT + (size_t)(tn * 16 + r16) * KDIM);

  v8f c = {};
  for (int kc = 0; kc < KDIM / 32; ++kc) {
    union { unsigned u[8]; v16bf v; } a, bmat;
    int k0 = kc * 32;
    #pragma unroll
    for (int r = 0; r < 8; ++r) {
      // A 16x32 bf16: VGPR r holds K = (r<4 ? 2r : 16+2(r-4)) + 8h, +1
      int kba = k0 + ((r < 4) ? (2 * r) : (16 + 2 * (r - 4))) + 8 * h;
      a.u[r] = aRow[kba >> 1];
      // B 32x16 bf16: VGPR r holds K = (r<4 ? 2r : 16+2(r-4)) + 8h, +1
      int kbb = k0 + ((r < 4) ? (2 * r) : (16 + 2 * (r - 4))) + 8 * h;
      bmat.u[r] = bRow[kbb >> 1];
    }
    c = __builtin_amdgcn_wmma_f32_16x16x32_bf16(false, a.v, false, bmat.v,
                                                (short)0, c, false, false);
  }

  // epilogue: D VGPR r -> tile row r + 8h; col = r16. Add restart term.
  int col = tn * 16 + r16;
  #pragma unroll
  for (int r = 0; r < 8; ++r) {
    int m = tm * 16 + r + 8 * h;          // m = ((e*256+i)*4 + u)
    int e = m >> 10, i = (m >> 2) & 255, u = m & 3;
    float v = c[r];
    if (u == e && col == i) v += back[e * NN + i] * 0.999f;   // (1 - BACKTRACK_FAILS_PROB)
    dist[(size_t)m * NN + col] = v;
  }
}

extern "C" void kernel_launch(void* const* d_in, const int* in_sizes, int n_in,
                              void* d_out, int out_size, void* d_ws, size_t ws_size,
                              hipStream_t stream) {
  (void)in_sizes; (void)n_in; (void)out_size; (void)ws_size;
  const float* vw     = (const float*)d_in[0];   // (256,256,4)
  const float* adj    = (const float*)d_in[1];   // (3,256,256)
  const float* lrp    = (const float*)d_in[2];   // (4,8,4,15)
  const int*   ntypes = (const int*)d_in[3];     // (256,)
  // d_in[4] = steps (scalar, fixed 48 per setup_inputs)
  float* accept = (float*)d_out;                 // (2,256,256) f32

  char* ws = (char*)d_ws;
  constexpr size_t offAnT    = 0;                                  // 256*768*2   = 393216
  constexpr size_t offPolicy = offAnT    + (size_t)NN * KDIM * 2;  // 1920*4
  constexpr size_t offPM     = offPolicy + 1920 * 4;               // 12*65536*4*4? -> 48 planes
  constexpr size_t offPA     = offPM     + (size_t)48 * NN * NN * 4;
  constexpr size_t offPB     = offPA     + (size_t)SS * NN * NN * 4;
  constexpr size_t offDist   = offPB     + (size_t)SS * NN * NN * 4;
  constexpr size_t offTMP    = offDist   + (size_t)MROWS * NN * 4;
  constexpr size_t offBack   = offTMP    + (size_t)MROWS * KDIM * 2;
  // total ~20.3 MB

  unsigned short* AnT = (unsigned short*)(ws + offAnT);
  float* policy       = (float*)(ws + offPolicy);
  float* PM           = (float*)(ws + offPM);
  float* PA           = (float*)(ws + offPA);
  float* PB           = (float*)(ws + offPB);
  float* dist         = (float*)(ws + offDist);
  unsigned short* TMP = (unsigned short*)(ws + offTMP);
  float* back         = (float*)(ws + offBack);

  fsm_anorm <<<TT * NN, 256, 0, stream>>>(adj, AnT);
  fsm_policy<<<1, 128, 0, stream>>>(lrp, policy);
  fsm_mixed <<<NN, 256, 0, stream>>>(vw, policy, ntypes, PM, PA, PB);
  fsm_init  <<<(MROWS * NN) / 256, 256, 0, stream>>>(dist, accept);

  for (int st = 0; st < STEPS; ++st) {
    fsm_expand<<<EE * NN, 256, 0, stream>>>(dist, PM, PA, PB, accept, back, TMP);
    fsm_gemm  <<<(MROWS / 16) * (NN / 16), 32, 0, stream>>>(TMP, AnT, back, dist);
  }
  // final accept accumulation (TMP/back writes are discarded)
  fsm_expand<<<EE * NN, 256, 0, stream>>>(dist, PM, PA, PB, accept, back, TMP);
}
